// LocationSensitiveSoftAttention_12489764896805
// MI455X (gfx1250) — matrix operand order repaired and checked by reading, not compiled, not run on hardware
//
#include <hip/hip_runtime.h>

// ---------------------------------------------------------------------------
// LocationSensitiveSoftAttention for MI455X (gfx1250, wave32, WMMA bf16)
// B=64, TQ=2, T=2048, HID=1024, ENC=512, U=256, FILT=32, K=31
// ---------------------------------------------------------------------------

#define B_   64
#define TQ_  2
#define T_   2048
#define HID_ 1024
#define ENC_ 512
#define U_   256
#define FILT_ 32
#define KW_  31

typedef __attribute__((ext_vector_type(8)))  float  v8f;
typedef __attribute__((ext_vector_type(8)))  __bf16 v8bf;
typedef __attribute__((ext_vector_type(16))) __bf16 v16bf;

__device__ __forceinline__ __bf16 f2bf(float f) {
    union { float f; unsigned int u; } c; c.f = f;
    unsigned int u = c.u;
    u += 0x7FFFu + ((u >> 16) & 1u);          // round-to-nearest-even
    unsigned short h = (unsigned short)(u >> 16);
    __bf16 r; __builtin_memcpy(&r, &h, 2);
    return r;
}
__device__ __forceinline__ float bf2f(__bf16 b) {
    unsigned short h; __builtin_memcpy(&h, &b, 2);
    union { unsigned int u; float f; } c; c.u = ((unsigned int)h) << 16;
    return c.f;
}

__device__ __forceinline__ v8f wmma_bf16(v16bf a, v16bf b, v8f c) {
    // (neg_a, A, neg_b, B, c_mod, C, reuse_a, reuse_b)
    return __builtin_amdgcn_wmma_f32_16x16x32_bf16(false, a, false, b, (short)0, c,
                                                   false, false);
}

// ---------------------------------------------------------------------------
// Prep kernels
// ---------------------------------------------------------------------------

// pq[b,u] = bq[u] + sum_h query[b,1,h] * Wq[h,u]   (only t=1 slice is used)
__global__ void prep_pq(const float* __restrict__ query,
                        const float* __restrict__ Wq,
                        const float* __restrict__ bq,
                        float* __restrict__ pq) {
    int b = blockIdx.x, u = threadIdx.x;
    const float* q = query + ((size_t)b * TQ_ + 1) * HID_;
    float acc = bq[u];
    for (int h = 0; h < HID_; ++h) acc += q[h] * Wq[h * U_ + u];
    pq[b * U_ + u] = acc;
}

// Weff[k,u] = sum_c conv_w[c,0,k]*Wl[c,u];  bias_loc[u] = bl[u] + sum_c conv_b[c]*Wl[c,u]
__global__ void prep_weff(const float* __restrict__ conv_w,
                          const float* __restrict__ conv_b,
                          const float* __restrict__ Wl,
                          const float* __restrict__ bl,
                          float* __restrict__ weff,
                          float* __restrict__ bias_loc) {
    int k = blockIdx.x, u = threadIdx.x;
    if (k < KW_) {
        float acc = 0.f;
        for (int c = 0; c < FILT_; ++c) acc += conv_w[c * KW_ + k] * Wl[c * U_ + u];
        weff[k * U_ + u] = acc;
    } else {
        float acc = bl[u];
        for (int c = 0; c < FILT_; ++c) acc += conv_b[c] * Wl[c * U_ + u];
        bias_loc[u] = acc;
    }
}

// WT[c,r] = bf16(W[r,c]); W is rows x cols row-major, WT is cols x rows
__global__ void prep_transpose_bf16(const float* __restrict__ W,
                                    __bf16* __restrict__ WT,
                                    int rows, int cols) {
    int i = blockIdx.x * 256 + threadIdx.x;
    if (i >= rows * cols) return;
    int c = i / rows, r = i % rows;
    WT[i] = f2bf(W[r * cols + c]);
}

// ---------------------------------------------------------------------------
// Main fused kernel: per block -> one batch b, 32 t-rows.
// 512 threads = 16 waves; wave w owns output columns [16w, 16w+16).
// GEMM1: mem_proj tile (bf16 WMMA, K=512)  -> store bf16 mem_proj
// epilogue: + bm + pq + folded conv loc    -> bf16 stage in LDS
// GEMM2: @ We (bf16 WMMA, K=256) -> tanh -> dot v_a -> sigmoid -> s
// ---------------------------------------------------------------------------

#define MROWS 32
#define MT    2     // M-tiles of 16 per wave

__global__ __launch_bounds__(512)
void attn_main(const float*  __restrict__ memory,
               const float*  __restrict__ state,
               const float*  __restrict__ pq,
               const float*  __restrict__ weff,
               const float*  __restrict__ bias_loc,
               const float*  __restrict__ bm,
               const __bf16* __restrict__ WmT,    // [U][ENC]
               const __bf16* __restrict__ WeT,    // [Uout][Uin]
               const float*  __restrict__ be,
               const float*  __restrict__ v_a,
               __bf16* __restrict__ memproj,      // [B][T][U]
               float*  __restrict__ sbuf)         // [B][T] sigmoid(energy)
{
    const int blk  = blockIdx.x;
    const int b    = blk >> 6;                // T/MROWS = 64 tiles per batch
    const int t0   = (blk & 63) * MROWS;
    const int tid  = threadIdx.x;
    const int wave = tid >> 5;
    const int lane = tid & 31;
    const int ln   = lane & 15;               // N column / A row within tile
    const int kg   = lane >> 4;               // K sub-group (0 or 1)

    __shared__ alignas(16) __bf16 Atile[MROWS * 520];   // 32 x (512+8) bf16
    __shared__ alignas(16) __bf16 Ptile[MROWS * 264];   // 32 x (256+8) bf16
    __shared__ float stateWin[64];                      // state[t0-15 .. t0+46]
    __shared__ float epart[16][MROWS];                  // per-wave energy partials

    // ---- cooperative load: memory tile f32 -> bf16 LDS (A-matrix staging) ----
    const float* msrc = memory + ((size_t)b * T_ + t0) * ENC_;
    for (int i = tid; i < MROWS * ENC_ / 4; i += 512) {
        float4 v = ((const float4*)msrc)[i];
        int e   = i * 4;
        int row = e >> 9;          // /ENC_
        int col = e & (ENC_ - 1);
        __bf16* d = &Atile[row * 520 + col];
        d[0] = f2bf(v.x); d[1] = f2bf(v.y); d[2] = f2bf(v.z); d[3] = f2bf(v.w);
    }
    if (tid < 64) {
        int t = t0 - 15 + tid;
        stateWin[tid] = (tid < MROWS + KW_ - 1 && t >= 0 && t < T_)
                            ? state[(size_t)b * T_ + t] : 0.f;
    }
    __syncthreads();

    const int u0   = wave * 16;
    const int ucol = u0 + ln;

    // ---------------- GEMM1: mem_proj = memory @ Wm ----------------
    v8f acc[MT];
    #pragma unroll
    for (int mt = 0; mt < MT; ++mt) acc[mt] = (v8f){0.f,0.f,0.f,0.f,0.f,0.f,0.f,0.f};

    const __bf16* Bb = WmT + (size_t)ucol * ENC_;
    #pragma unroll
    for (int kk = 0; kk < ENC_ / 32; ++kk) {
        const int k0 = kk * 32;
        v16bf bmat = *(const v16bf*)(Bb + k0 + kg * 16);
        #pragma unroll
        for (int mt = 0; mt < MT; ++mt) {
            const __bf16* ap = &Atile[(mt * 16 + ln) * 520 + k0 + kg * 16];
            v8bf lo = *(const v8bf*)(ap);
            v8bf hi = *(const v8bf*)(ap + 8);
            v16bf a = __builtin_shufflevector(lo, hi,
                        0,1,2,3,4,5,6,7,8,9,10,11,12,13,14,15);
            acc[mt] = wmma_bf16(a, bmat, acc[mt]);
        }
    }

    // ---------------- epilogue 1: loc filter + bias folds ----------------
    const float bmv = bm[ucol];
    const float pqv = pq[b * U_ + ucol];
    const float blv = bias_loc[ucol];

    float loc[MT][8];
    #pragma unroll
    for (int mt = 0; mt < MT; ++mt)
        #pragma unroll
        for (int r = 0; r < 8; ++r) loc[mt][r] = blv;

    #pragma unroll
    for (int k = 0; k < KW_; ++k) {
        const float w = weff[k * U_ + ucol];
        #pragma unroll
        for (int mt = 0; mt < MT; ++mt)
            #pragma unroll
            for (int r = 0; r < 8; ++r)
                loc[mt][r] += stateWin[mt * 16 + r + 8 * kg + k] * w;
    }

    __bf16* mpb = memproj + ((size_t)b * T_ + t0) * U_ + ucol;
    #pragma unroll
    for (int mt = 0; mt < MT; ++mt) {
        #pragma unroll
        for (int r = 0; r < 8; ++r) {
            const int row = mt * 16 + r + 8 * kg;       // local t row
            const float mp = acc[mt][r] + bmv;          // mem_proj value
            mpb[(size_t)row * U_] = f2bf(mp);
            const float pre = mp + pqv + loc[mt][r];
            Ptile[row * 264 + ucol] = f2bf(pre);
        }
    }
    __syncthreads();

    // ---------------- GEMM2: h = tanh(pre @ We + be) ----------------
    v8f acc2[MT];
    #pragma unroll
    for (int mt = 0; mt < MT; ++mt) acc2[mt] = (v8f){0.f,0.f,0.f,0.f,0.f,0.f,0.f,0.f};

    const __bf16* B2 = WeT + (size_t)ucol * U_;
    #pragma unroll
    for (int kk = 0; kk < U_ / 32; ++kk) {
        const int k0 = kk * 32;
        v16bf bmat = *(const v16bf*)(B2 + k0 + kg * 16);
        #pragma unroll
        for (int mt = 0; mt < MT; ++mt) {
            const __bf16* ap = &Ptile[(mt * 16 + ln) * 264 + k0 + kg * 16];
            v8bf lo = *(const v8bf*)(ap);
            v8bf hi = *(const v8bf*)(ap + 8);
            v16bf a = __builtin_shufflevector(lo, hi,
                        0,1,2,3,4,5,6,7,8,9,10,11,12,13,14,15);
            acc2[mt] = wmma_bf16(a, bmat, acc2[mt]);
        }
    }

    // ---------------- energy = tanh(. + be) . v_a, deterministic reduce ----
    const float bev = be[ucol];
    const float vav = v_a[ucol];
    #pragma unroll
    for (int mt = 0; mt < MT; ++mt) {
        #pragma unroll
        for (int r = 0; r < 8; ++r) {
            float e = tanhf(acc2[mt][r] + bev) * vav;
            e += __shfl_xor(e, 1);
            e += __shfl_xor(e, 2);
            e += __shfl_xor(e, 4);
            e += __shfl_xor(e, 8);
            if (ln == 0) epart[wave][mt * 16 + r + 8 * kg] = e;
        }
    }
    __syncthreads();

    if (tid < MROWS) {
        float e = 0.f;
        #pragma unroll
        for (int w = 0; w < 16; ++w) e += epart[w][tid];
        const float sg = 1.f / (1.f + expf(-e));
        sbuf[(size_t)b * T_ + t0 + tid] = sg;
    }
}

// ---------------------------------------------------------------------------
// Post kernels
// ---------------------------------------------------------------------------

__global__ void reduce_s(const float* __restrict__ sbuf, float* __restrict__ srow) {
    int b = blockIdx.x, tid = threadIdx.x;
    __shared__ float red[256];
    float acc = 0.f;
    for (int t = tid; t < T_; t += 256) acc += sbuf[(size_t)b * T_ + t];
    red[tid] = acc; __syncthreads();
    for (int s = 128; s > 0; s >>= 1) {
        if (tid < s) red[tid] += red[tid + s];
        __syncthreads();
    }
    if (tid == 0) srow[b] = red[0];
}

// partial context over a 256-wide t-chunk (deterministic; no atomics)
__global__ void ctx_part(const float* __restrict__ state,
                         const float* __restrict__ sbuf,
                         const float* __restrict__ srow,
                         const __bf16* __restrict__ memproj,
                         float* __restrict__ ctx) {
    int b = blockIdx.x >> 3, ch = blockIdx.x & 7, tid = threadIdx.x;
    __shared__ float wbuf[256];
    const int t0 = ch * 256;
    const float inv = 1.f / srow[b];
    wbuf[tid] = state[(size_t)b * T_ + t0 + tid] + sbuf[(size_t)b * T_ + t0 + tid] * inv;
    __syncthreads();
    const __bf16* mp = memproj + ((size_t)b * T_ + t0) * U_ + tid;
    float acc = 0.f;
    for (int t = 0; t < 256; ++t) acc += wbuf[t] * bf2f(mp[(size_t)t * U_]);
    ctx[((size_t)ch * B_ + b) * U_ + tid] = acc;
}

__global__ void ctx_final(const float* __restrict__ ctx, float* __restrict__ out) {
    int b = blockIdx.x, u = threadIdx.x;
    float acc = 0.f;
    #pragma unroll
    for (int ch = 0; ch < 8; ++ch) acc += ctx[((size_t)ch * B_ + b) * U_ + u];
    out[b * U_ + u] = acc;   // output shape (B,1,U)
}

// ---------------------------------------------------------------------------
// Launch
// ---------------------------------------------------------------------------

extern "C" void kernel_launch(void* const* d_in, const int* in_sizes, int n_in,
                              void* d_out, int out_size, void* d_ws, size_t ws_size,
                              hipStream_t stream) {
    (void)in_sizes; (void)n_in; (void)out_size; (void)ws_size;

    const float* query  = (const float*)d_in[0];
    const float* state  = (const float*)d_in[1];
    const float* memory = (const float*)d_in[2];
    const float* Wq     = (const float*)d_in[3];
    const float* bq     = (const float*)d_in[4];
    const float* Wm     = (const float*)d_in[5];
    const float* bm     = (const float*)d_in[6];
    const float* Wl     = (const float*)d_in[7];
    const float* bl     = (const float*)d_in[8];
    const float* conv_w = (const float*)d_in[9];
    const float* conv_b = (const float*)d_in[10];
    const float* We     = (const float*)d_in[11];
    const float* be     = (const float*)d_in[12];
    const float* v_a    = (const float*)d_in[13];
    float* out = (float*)d_out;

    char* w = (char*)d_ws;
    float*  pq       = (float*)(w + 0);          //  65536 B
    float*  weff     = (float*)(w + 65536);      //  31744 B (pad to 32768)
    float*  bias_loc = (float*)(w + 98304);      //   1024 B
    __bf16* WmT      = (__bf16*)(w + 99328);     // 262144 B  (U x ENC)
    __bf16* WeT      = (__bf16*)(w + 361472);    // 131072 B  (U x U)
    float*  sbuf     = (float*)(w + 492544);     // 524288 B  (B x T)
    float*  srow     = (float*)(w + 1016832);    //    256 B
    float*  ctx      = (float*)(w + 1017088);    // 524288 B  (8 x B x U)
    __bf16* memproj  = (__bf16*)(w + 1541376);   // 67108864 B (B x T x U)

    prep_pq<<<B_, U_, 0, stream>>>(query, Wq, bq, pq);
    prep_weff<<<KW_ + 1, U_, 0, stream>>>(conv_w, conv_b, Wl, bl, weff, bias_loc);
    prep_transpose_bf16<<<(U_ * ENC_) / 256, 256, 0, stream>>>(Wm, WmT, ENC_, U_);
    prep_transpose_bf16<<<(U_ * U_) / 256, 256, 0, stream>>>(We, WeT, U_, U_);

    attn_main<<<B_ * (T_ / MROWS), 512, 0, stream>>>(
        memory, state, pq, weff, bias_loc, bm, WmT, WeT, be, v_a, memproj, sbuf);

    reduce_s<<<B_, 256, 0, stream>>>(sbuf, srow);
    ctx_part<<<B_ * 8, 256, 0, stream>>>(state, sbuf, srow, memproj, ctx);
    ctx_final<<<B_, U_, 0, stream>>>(ctx, out);
}
